// TorchLMHeadCE_3100966388352
// MI455X (gfx1250) — matrix operand order repaired and checked by reading, not compiled, not run on hardware
//
#include <hip/hip_runtime.h>
#include <hip/hip_bf16.h>

// ---------------- problem constants ----------------
#define N_ROWS   4096
#define H_DIM    2048
#define V_DIM    50257
#define IGNORE_INDEX (-100)

// ---------------- tiling ----------------
#define VT       128          // W rows (logit cols) per workgroup
#define NT       64           // x rows per n-iteration
#define KT       32           // K chunk staged in LDS (double buffered)
#define NCH      (H_DIM / KT) // 64 chunks
#define VB       393          // ceil(V/VT)
#define NSPLIT   8            // grid.y: splits of the n-loop
#define NBPER    (N_ROWS / NT / NSPLIT)   // n-tiles per block = 8
#define LDST     40           // padded LDS row stride (ushorts): conflict-free frag loads

typedef __bf16 bf16_t;
typedef bf16_t v16bf __attribute__((ext_vector_type(16)));
typedef float  v8f   __attribute__((ext_vector_type(8)));

union Frag {
    uint4 u[2];
    v16bf v;
};

// Truncating fp32 -> bf16 hi/lo split for a pair of floats, packed into
// (hi1|hi0) and (lo1|lo0) with v_perm_b32.  hi = top 16 bits of f (round
// toward zero), lo = bf16(f - hi): the lo term absorbs the truncation error,
// total representation error ~2^-16 relative.  3 VALU per float.
__device__ __forceinline__ void cvt2(float f0, float f1, unsigned& hiP, unsigned& loP) {
    unsigned u0 = __float_as_uint(f0), u1 = __float_as_uint(f1);
    float h0 = __uint_as_float(u0 & 0xFFFF0000u);
    float h1 = __uint_as_float(u1 & 0xFFFF0000u);
    unsigned d0 = __float_as_uint(f0 - h0);
    unsigned d1 = __float_as_uint(f1 - h1);
#if __has_builtin(__builtin_amdgcn_perm)
    hiP = __builtin_amdgcn_perm(u1, u0, 0x07060302u);
    loP = __builtin_amdgcn_perm(d1, d0, 0x07060302u);
#else
    hiP = (u0 >> 16) | (u1 & 0xFFFF0000u);
    loP = (d0 >> 16) | (d1 & 0xFFFF0000u);
#endif
}

__global__ __launch_bounds__(256)
void lmhead_partial_kernel(const float* __restrict__ x,
                           const float* __restrict__ W,
                           const int*   __restrict__ y,
                           float* __restrict__ pmax,
                           float* __restrict__ psum,
                           float* __restrict__ tgt)
{
    __shared__ __align__(16) unsigned short xs_hi[2][NT * LDST];
    __shared__ __align__(16) unsigned short xs_lo[2][NT * LDST];
    __shared__ __align__(16) unsigned short wb_hi[2][VT * LDST];
    __shared__ __align__(16) unsigned short wb_lo[2][VT * LDST];
    __shared__ float redbuf[NT * 4];
    __shared__ float redrow[NT];
    __shared__ int   ytile[NT];

    const int tid    = threadIdx.x;
    const int lane   = tid & 31;
    const int waveId = tid >> 5;
    const int wm     = waveId & 1;   // 2 wave-rows over NT=64
    const int wn     = waveId >> 1;  // 4 wave-cols over VT=128
    const int lm     = lane & 15;
    const int half   = lane >> 4;

    const int vb    = blockIdx.x;
    const int vbase = vb * VT;

    // staging assignments (per K chunk of 32 columns)
    const int xrow = tid >> 2;        // x: 64 rows, 4 threads/row, 8 floats (2 float4)
    const int xseg = tid & 3;
    const int wrow = tid >> 1;        // W: 128 rows, 2 threads/row, 16 floats (4 float4)
    const int wseg = tid & 1;
    int wg = vbase + wrow; if (wg > V_DIM - 1) wg = V_DIM - 1;
    const float* gw = W + (size_t)wg * H_DIM + wseg * 16;

    // fragment column geometry
    const int n0  = wn * 32 + lm;
    const int n1  = n0 + 16;
    const int v0g = vbase + n0;
    const int v1g = vbase + n1;
    const bool val0 = (v0g < V_DIM);
    const bool val1 = (v1g < V_DIM);

    float4 rx[2], rw[4];

    for (int ib = 0; ib < NBPER; ++ib) {
        const int nb = blockIdx.y * NBPER + ib;
        const int nrowbase = nb * NT;
        const float* gx = x + (size_t)(nrowbase + xrow) * H_DIM + xseg * 8;

        if (tid < NT) ytile[tid] = y[nrowbase + tid];

        v8f acc[2][2];
        #pragma unroll
        for (int mt = 0; mt < 2; ++mt)
            #pragma unroll
            for (int nt = 0; nt < 2; ++nt)
                #pragma unroll
                for (int r = 0; r < 8; ++r)
                    acc[mt][nt][r] = 0.0f;

        // ---- prologue: stage chunk 0 into buffer 0 ----
        #pragma unroll
        for (int i = 0; i < 2; ++i) rx[i] = ((const float4*)gx)[i];
        #pragma unroll
        for (int i = 0; i < 4; ++i) rw[i] = ((const float4*)gw)[i];
        {
            unsigned short* dxh = &xs_hi[0][xrow * LDST + xseg * 8];
            unsigned short* dxl = &xs_lo[0][xrow * LDST + xseg * 8];
            #pragma unroll
            for (int i = 0; i < 2; ++i) {
                unsigned h01, l01, h23, l23;
                cvt2(rx[i].x, rx[i].y, h01, l01);
                cvt2(rx[i].z, rx[i].w, h23, l23);
                uint2 th; th.x = h01; th.y = h23;
                uint2 tl; tl.x = l01; tl.y = l23;
                *(uint2*)(dxh + i * 4) = th;
                *(uint2*)(dxl + i * 4) = tl;
            }
            unsigned short* dwh = &wb_hi[0][wrow * LDST + wseg * 16];
            unsigned short* dwl = &wb_lo[0][wrow * LDST + wseg * 16];
            #pragma unroll
            for (int i = 0; i < 4; ++i) {
                unsigned h01, l01, h23, l23;
                cvt2(rw[i].x, rw[i].y, h01, l01);
                cvt2(rw[i].z, rw[i].w, h23, l23);
                uint2 th; th.x = h01; th.y = h23;
                uint2 tl; tl.x = l01; tl.y = l23;
                *(uint2*)(dwh + i * 4) = th;
                *(uint2*)(dwl + i * 4) = tl;
            }
        }

        // ---- software-pipelined main loop ----
        for (int kc = 0; kc < NCH; ++kc) {
            const int cur = kc & 1;
            const int nxt = cur ^ 1;
            const bool more = (kc + 1 < NCH);

            // issue global loads for the next chunk (latency hidden under WMMA)
            if (more) {
                const int ko = (kc + 1) * KT;
                #pragma unroll
                for (int i = 0; i < 2; ++i) rx[i] = ((const float4*)(gx + ko))[i];
                #pragma unroll
                for (int i = 0; i < 4; ++i) rw[i] = ((const float4*)(gw + ko))[i];
            }

            __syncthreads();   // buf[cur] stores visible; prior reads of buf[nxt] done

            // ---- WMMA on buf[cur] ----
            Frag ah[2], al[2], bh[2], bl[2];
            #pragma unroll
            for (int mt = 0; mt < 2; ++mt) {
                int row  = wm * 32 + mt * 16 + lm;
                int base = row * LDST + 8 * half;
                ah[mt].u[0] = *(const uint4*)&xs_hi[cur][base];
                ah[mt].u[1] = *(const uint4*)&xs_hi[cur][base + 16];
                al[mt].u[0] = *(const uint4*)&xs_lo[cur][base];
                al[mt].u[1] = *(const uint4*)&xs_lo[cur][base + 16];
            }
            #pragma unroll
            for (int nt = 0; nt < 2; ++nt) {
                int vr   = wn * 32 + nt * 16 + lm;
                int base = vr * LDST + 8 * half;
                bh[nt].u[0] = *(const uint4*)&wb_hi[cur][base];
                bh[nt].u[1] = *(const uint4*)&wb_hi[cur][base + 16];
                bl[nt].u[0] = *(const uint4*)&wb_lo[cur][base];
                bl[nt].u[1] = *(const uint4*)&wb_lo[cur][base + 16];
            }
            #pragma unroll
            for (int mt = 0; mt < 2; ++mt)
                #pragma unroll
                for (int nt = 0; nt < 2; ++nt) {
                    acc[mt][nt] = __builtin_amdgcn_wmma_f32_16x16x32_bf16(
                        false, ah[mt].v, false, bh[nt].v, (short)0, acc[mt][nt], false, false);
                    acc[mt][nt] = __builtin_amdgcn_wmma_f32_16x16x32_bf16(
                        false, ah[mt].v, false, bl[nt].v, (short)0, acc[mt][nt], false, false);
                    acc[mt][nt] = __builtin_amdgcn_wmma_f32_16x16x32_bf16(
                        false, al[mt].v, false, bh[nt].v, (short)0, acc[mt][nt], false, false);
                }

            // ---- convert + store next chunk into buf[nxt]; co-executes with WMMA ----
            if (more) {
                unsigned short* dxh = &xs_hi[nxt][xrow * LDST + xseg * 8];
                unsigned short* dxl = &xs_lo[nxt][xrow * LDST + xseg * 8];
                #pragma unroll
                for (int i = 0; i < 2; ++i) {
                    unsigned h01, l01, h23, l23;
                    cvt2(rx[i].x, rx[i].y, h01, l01);
                    cvt2(rx[i].z, rx[i].w, h23, l23);
                    uint2 th; th.x = h01; th.y = h23;
                    uint2 tl; tl.x = l01; tl.y = l23;
                    *(uint2*)(dxh + i * 4) = th;
                    *(uint2*)(dxl + i * 4) = tl;
                }
                unsigned short* dwh = &wb_hi[nxt][wrow * LDST + wseg * 16];
                unsigned short* dwl = &wb_lo[nxt][wrow * LDST + wseg * 16];
                #pragma unroll
                for (int i = 0; i < 4; ++i) {
                    unsigned h01, l01, h23, l23;
                    cvt2(rw[i].x, rw[i].y, h01, l01);
                    cvt2(rw[i].z, rw[i].w, h23, l23);
                    uint2 th; th.x = h01; th.y = h23;
                    uint2 tl; tl.x = l01; tl.y = l23;
                    *(uint2*)(dwh + i * 4) = th;
                    *(uint2*)(dwl + i * 4) = tl;
                }
            }
        }

        // ---------------- epilogue: partial softmax stats ----------------
        // D layout: (M = r + 8*half + 16*mt + 32*wm, N = lm + 16*nt + 32*wn) = acc[mt][nt][r]
        float mx[2][8];
        #pragma unroll
        for (int mt = 0; mt < 2; ++mt) {
            #pragma unroll
            for (int r = 0; r < 8; ++r) {
                int mloc = wm * 32 + mt * 16 + r + 8 * half;
                float a0 = acc[mt][0][r];
                float a1 = acc[mt][1][r];
                int t = ytile[mloc];
                if (val0 && t == v0g) tgt[nrowbase + mloc] = a0;
                if (val1 && t == v1g) tgt[nrowbase + mloc] = a1;
                float m = fmaxf(val0 ? a0 : -3.0e38f, val1 ? a1 : -3.0e38f);
                #pragma unroll
                for (int s = 1; s < 16; s <<= 1)
                    m = fmaxf(m, __shfl_xor(m, s, 32));
                mx[mt][r] = m;
            }
        }
        if (lm == 0) {
            #pragma unroll
            for (int mt = 0; mt < 2; ++mt)
                #pragma unroll
                for (int r = 0; r < 8; ++r)
                    redbuf[(wm * 32 + mt * 16 + r + 8 * half) * 4 + wn] = mx[mt][r];
        }
        __syncthreads();
        if (tid < NT) {
            float m = fmaxf(fmaxf(redbuf[tid * 4], redbuf[tid * 4 + 1]),
                            fmaxf(redbuf[tid * 4 + 2], redbuf[tid * 4 + 3]));
            redrow[tid] = m;
        }
        __syncthreads();

        float sm[2][8];
        #pragma unroll
        for (int mt = 0; mt < 2; ++mt) {
            #pragma unroll
            for (int r = 0; r < 8; ++r) {
                int mloc = wm * 32 + mt * 16 + r + 8 * half;
                float M = redrow[mloc];
                float s = (val0 ? __expf(acc[mt][0][r] - M) : 0.0f)
                        + (val1 ? __expf(acc[mt][1][r] - M) : 0.0f);
                #pragma unroll
                for (int d = 1; d < 16; d <<= 1)
                    s += __shfl_xor(s, d, 32);
                sm[mt][r] = s;
            }
        }
        if (lm == 0) {
            #pragma unroll
            for (int mt = 0; mt < 2; ++mt)
                #pragma unroll
                for (int r = 0; r < 8; ++r)
                    redbuf[(wm * 32 + mt * 16 + r + 8 * half) * 4 + wn] = sm[mt][r];
        }
        __syncthreads();
        if (tid < NT) {
            float S = redbuf[tid * 4] + redbuf[tid * 4 + 1]
                    + redbuf[tid * 4 + 2] + redbuf[tid * 4 + 3];
            size_t grow = (size_t)(nrowbase + tid);
            pmax[grow * VB + vb] = redrow[tid];
            psum[grow * VB + vb] = S;
        }
        __syncthreads();
    }
}

__global__ void row_reduce_kernel(const float* __restrict__ pmax,
                                  const float* __restrict__ psum,
                                  const float* __restrict__ tgt,
                                  const int*   __restrict__ y,
                                  float* __restrict__ nll)
{
    int n = blockIdx.x * blockDim.x + threadIdx.x;
    if (n >= N_ROWS) return;
    const float* pm = pmax + (size_t)n * VB;
    const float* ps = psum + (size_t)n * VB;
    float M = -3.0e38f;
    for (int i = 0; i < VB; ++i) M = fmaxf(M, pm[i]);
    float S = 0.0f;
    for (int i = 0; i < VB; ++i) S += ps[i] * __expf(pm[i] - M);
    int t = y[n];
    nll[n] = (t != IGNORE_INDEX) ? (M + logf(S) - tgt[n]) : 0.0f;
}

__global__ void finalize_kernel(const float* __restrict__ nll,
                                const int*   __restrict__ y,
                                float* __restrict__ out)
{
    __shared__ float ssum[256];
    __shared__ int   scnt[256];
    int tid = threadIdx.x;
    float s = 0.0f; int c = 0;
    for (int i = tid; i < N_ROWS; i += 256) {
        s += nll[i];
        c += (y[i] != IGNORE_INDEX) ? 1 : 0;
    }
    ssum[tid] = s; scnt[tid] = c;
    __syncthreads();
    for (int o = 128; o > 0; o >>= 1) {
        if (tid < o) { ssum[tid] += ssum[tid + o]; scnt[tid] += scnt[tid + o]; }
        __syncthreads();
    }
    if (tid == 0) {
        int cv = scnt[0] > 0 ? scnt[0] : 1;
        out[0] = ssum[0] / (float)cv;
    }
}

extern "C" void kernel_launch(void* const* d_in, const int* in_sizes, int n_in,
                              void* d_out, int out_size, void* d_ws, size_t ws_size,
                              hipStream_t stream) {
    const float* x = (const float*)d_in[0];
    const int*   y = (const int*)d_in[1];
    const float* W = (const float*)d_in[2];
    float* out = (float*)d_out;

    char* ws = (char*)d_ws;
    size_t off = 0;
    float* pmax = (float*)(ws + off); off += (size_t)N_ROWS * VB * sizeof(float);
    float* psum = (float*)(ws + off); off += (size_t)N_ROWS * VB * sizeof(float);
    float* tgt  = (float*)(ws + off); off += (size_t)N_ROWS * sizeof(float);
    float* nll  = (float*)(ws + off); off += (size_t)N_ROWS * sizeof(float);

    dim3 grid(VB, NSPLIT);
    lmhead_partial_kernel<<<grid, 256, 0, stream>>>(x, W, y, pmax, psum, tgt);
    row_reduce_kernel<<<N_ROWS / 256, 256, 0, stream>>>(pmax, psum, tgt, y, nll);
    finalize_kernel<<<1, 256, 0, stream>>>(nll, y, out);
}
